// ParticleTransformer_58334245814894
// MI455X (gfx1250) — compile-verified, matrix operand order checked by reading, not run on hardware
//
#include <hip/hip_runtime.h>
#include <hip/hip_bf16.h>

typedef __bf16 bf16;
typedef __bf16 bf16x8  __attribute__((ext_vector_type(8)));
typedef __bf16 bf16x16 __attribute__((ext_vector_type(16)));
typedef float  f32x8   __attribute__((ext_vector_type(8)));

#define LN_EPS 1e-5f

__device__ __forceinline__ float geluf(float x) {
  return 0.5f * x * (1.0f + erff(x * 0.7071067811865476f));
}

__device__ __forceinline__ float wave_sum32(float v) {
#pragma unroll
  for (int off = 16; off > 0; off >>= 1) v += __shfl_xor(v, off, 32);
  return v;
}

__device__ __forceinline__ bf16x16 load_frag16(const bf16* p) {
  bf16x8 lo = *(const bf16x8*)(p);
  bf16x8 hi = *(const bf16x8*)(p + 16);
  return __builtin_shufflevector(lo, hi, 0,1,2,3,4,5,6,7,8,9,10,11,12,13,14,15);
}

// ---------------------------------------------------------------------------
// GEMM v2: Y(M x O, f32) = A(M x K, bf16) * W(O x K, bf16)^T + bias
// Block = 256 threads (8 waves), tile = 128 rows x (16*NBN) cols.
// Each wave: 16 x (16*NBN) strip -> NBN accumulators; one A fragment is
// reused across NBN v_wmma_f32_16x16x32_bf16 per K-chunk. W chunk for the
// block's columns is cooperatively staged in double-buffered LDS and the
// B fragments are read with ds_load_b128 (W traffic /8, A-frag traffic /NBN).
// Requires: M % 128 == 0, K % 32 == 0, O % (16*NBN) == 0 (exact grids).
// ---------------------------------------------------------------------------
template <int NBN>
__global__ __launch_bounds__(256) void gemm_bf16_wmma(
    const bf16* __restrict__ A, const bf16* __restrict__ W,
    const float* __restrict__ bias, float* __restrict__ Y,
    int M, int K, int O)
{
  constexpr int BN = 16 * NBN;                 // block cols
  __shared__ __align__(16) bf16 wtile[2][BN * 32];
  int tilesN = O / BN;
  int bm = blockIdx.x / tilesN;
  int bn = blockIdx.x % tilesN;
  int tid  = threadIdx.x;
  int wid  = tid >> 5;                         // 0..7  -> 16-row strip
  int lane = tid & 31;
  int r    = lane & 15;
  int kb   = (lane & 16) ? 8 : 0;              // k sub-offset per fragment layout
  const bf16* ap = A + (size_t)(bm * 128 + wid * 16 + r) * K + kb;

  // cooperative W-chunk loader: BN*32 bf16 per chunk, 16B per thread
  int lrow = (tid * 8) / 32;                   // W row within tile
  int lcol = (tid * 8) % 32;                   // k within chunk
  const bf16* wsrc = W + (size_t)(bn * BN + lrow) * K + lcol;

  f32x8 zero = {0.f,0.f,0.f,0.f,0.f,0.f,0.f,0.f};
  f32x8 acc[NBN];
#pragma unroll
  for (int t = 0; t < NBN; t++) acc[t] = zero;

  if (tid < BN * 4)
    *(bf16x8*)&wtile[0][tid * 8] = *(const bf16x8*)(wsrc);
  __syncthreads();

  int buf = 0;
  for (int k0 = 0; k0 < K; k0 += 32) {
    int nb = buf ^ 1;
    if (k0 + 32 < K && tid < BN * 4)           // stage next chunk (other buffer)
      *(bf16x8*)&wtile[nb][tid * 8] = *(const bf16x8*)(wsrc + k0 + 32);
    if (k0 + 128 < K)
      __builtin_prefetch(ap + k0 + 128, 0, 0); // global_prefetch_b8
    bf16x16 fa = load_frag16(ap + k0);
#pragma unroll
    for (int t = 0; t < NBN; t++) {
      bf16x16 fb = load_frag16(&wtile[buf][(t * 16 + r) * 32 + kb]);
      acc[t] = __builtin_amdgcn_wmma_f32_16x16x32_bf16(false, fa, false, fb,
                                                       (short)0, acc[t], false, false);
    }
    __syncthreads();
    buf = nb;
  }

  int rowb = bm * 128 + wid * 16 + ((lane & 16) ? 8 : 0);  // C/D layout
#pragma unroll
  for (int t = 0; t < NBN; t++) {
    int col = bn * BN + t * 16 + r;
    float bv = bias ? bias[col] : 0.f;
#pragma unroll
    for (int v = 0; v < 8; v++)
      Y[(size_t)(rowb + v) * O + col] = acc[t][v] + bv;
  }
}

// ---------------------------------------------------------------------------
// Pack f32 weights (O x Kw) -> bf16 (O x Kp), zero-padded K (also used to pad
// activations like the 4-feature interactions to K=32).
// ---------------------------------------------------------------------------
__global__ __launch_bounds__(256) void pack_w_kernel(
    const float* __restrict__ W, int O, int Kw, int Kp, bf16* __restrict__ out)
{
  int tid = blockIdx.x * blockDim.x + threadIdx.x;
  if (tid >= O * Kp) return;
  int c = tid % Kp;
  int o = tid / Kp;
  out[tid] = (bf16)((c < Kw) ? W[(size_t)o * Kw + c] : 0.f);
}

// ---------------------------------------------------------------------------
// kNN over coords x[..., 8:10]; k smallest d2, index tiebreak (matches top_k).
// ---------------------------------------------------------------------------
__global__ __launch_bounds__(128) void knn_kernel(
    const float* __restrict__ x, int C, int kk, int* __restrict__ idx)
{
  int b = blockIdx.x;
  int n = threadIdx.x;             // 128 particles
  __shared__ float cx[128], cy[128];
  cx[n] = x[((size_t)b * 128 + n) * C + 8];
  cy[n] = x[((size_t)b * 128 + n) * C + 9];
  __syncthreads();
  unsigned mask[4] = {0u, 0u, 0u, 0u};
  float qx = cx[n], qy = cy[n];
  for (int t = 0; t < kk; t++) {
    float best = 3.4e38f;
    int bi = 0;
    for (int m = 0; m < 128; m++) {
      if ((mask[m >> 5] >> (m & 31)) & 1u) continue;
      float dx = qx - cx[m], dy = qy - cy[m];
      float d2 = dx * dx + dy * dy;
      if (d2 < best) { best = d2; bi = m; }
    }
    mask[bi >> 5] |= 1u << (bi & 31);
    idx[((size_t)b * 128 + n) * kk + t] = bi;
  }
}

// Build EdgeConv input rows: [ctr(C), nb-ctr(C)] padded to Kp, bf16.
__global__ __launch_bounds__(256) void edge_build_kernel(
    const float* __restrict__ x, const int* __restrict__ idx,
    int C, int kk, int Kp, bf16* __restrict__ A)
{
  size_t tid = (size_t)blockIdx.x * blockDim.x + threadIdx.x;
  size_t total = (size_t)4096 * kk * Kp;
  if (tid >= total) return;
  int c = (int)(tid % Kp);
  size_t row = tid / Kp;
  int j = (int)(row % kk);
  size_t pn = row / kk;                 // b*128 + n
  float v = 0.f;
  if (c < C) {
    v = x[pn * C + c];
  } else if (c < 2 * C) {
    int m = idx[pn * kk + j];
    size_t bb = (pn >> 7) << 7;
    v = x[(bb + m) * C + (c - C)] - x[pn * C + (c - C)];
  }
  A[tid] = (bf16)v;
}

// Mean over k neighbors: x[pn,c] = mean_j H[(pn*kk+j)*C + c]
__global__ __launch_bounds__(256) void mean_k_kernel(
    const bf16* __restrict__ H, int C, int kk, float* __restrict__ xout)
{
  int tid = blockIdx.x * blockDim.x + threadIdx.x;
  if (tid >= 4096 * C) return;
  int c = tid % C;
  int pn = tid / C;
  float s = 0.f;
  for (int j = 0; j < kk; j++) s += (float)H[((size_t)pn * kk + j) * C + c];
  xout[tid] = s / (float)kk;
}

// Row LayerNorm (+optional exact GELU), one wave per row, bf16 out.
__global__ __launch_bounds__(256) void ln_act_rows(
    const float* __restrict__ Y, const float* __restrict__ g,
    const float* __restrict__ be, bf16* __restrict__ out,
    int M, int C, int dogelu)
{
  int row = blockIdx.x * (blockDim.x >> 5) + (threadIdx.x >> 5);
  if (row >= M) return;
  int lane = threadIdx.x & 31;
  const float* r = Y + (size_t)row * C;
  float s = 0.f;
  for (int i = lane; i < C; i += 32) s += r[i];
  float m = wave_sum32(s) / C;
  float v = 0.f;
  for (int i = lane; i < C; i += 32) { float d = r[i] - m; v += d * d; }
  float inv = rsqrtf(wave_sum32(v) / C + LN_EPS);
  bf16* o = out + (size_t)row * C;
  for (int i = lane; i < C; i += 32) {
    float y = (r[i] - m) * inv * g[i] + be[i];
    o[i] = (bf16)(dogelu ? geluf(y) : y);
  }
}

// t = gelu(y); z = ln(t, g, b) -> bf16 (MLP mid, C up to 1024, re-read style)
__global__ __launch_bounds__(256) void gelu_ln_rows_kernel(
    const float* __restrict__ Y, const float* __restrict__ g,
    const float* __restrict__ be, bf16* __restrict__ out, int M, int C)
{
  int row = blockIdx.x * (blockDim.x >> 5) + (threadIdx.x >> 5);
  if (row >= M) return;
  int lane = threadIdx.x & 31;
  const float* r = Y + (size_t)row * C;
  float s = 0.f;
  for (int i = lane; i < C; i += 32) s += geluf(r[i]);
  float m = wave_sum32(s) / C;
  float v = 0.f;
  for (int i = lane; i < C; i += 32) { float d = geluf(r[i]) - m; v += d * d; }
  float inv = rsqrtf(wave_sum32(v) / C + LN_EPS);
  bf16* o = out + (size_t)row * C;
  for (int i = lane; i < C; i += 32)
    o[i] = (bf16)((geluf(r[i]) - m) * inv * g[i] + be[i]);
}

// h2 = ln(a, n2) + x (f32 out) ; z = ln(h2, g1, b1) (bf16 out). C = 256 fixed.
__global__ __launch_bounds__(256) void ln_res_ln_kernel(
    const float* __restrict__ Ain, const float* __restrict__ X,
    const float* __restrict__ n2g, const float* __restrict__ n2b,
    const float* __restrict__ g1, const float* __restrict__ b1,
    float* __restrict__ H2, bf16* __restrict__ Z, int M)
{
  const int C = 256;
  int row = blockIdx.x * (blockDim.x >> 5) + (threadIdx.x >> 5);
  if (row >= M) return;
  int lane = threadIdx.x & 31;
  const float* ar = Ain + (size_t)row * C;
  float t[8], hh[8];
  float s = 0.f;
#pragma unroll
  for (int j = 0; j < 8; j++) { t[j] = ar[lane + 32 * j]; s += t[j]; }
  float m = wave_sum32(s) * (1.f / 256.f);
  float v = 0.f;
#pragma unroll
  for (int j = 0; j < 8; j++) { float d = t[j] - m; v += d * d; }
  float inv = rsqrtf(wave_sum32(v) * (1.f / 256.f) + LN_EPS);
  const float* xr = X + (size_t)row * C;
  s = 0.f;
#pragma unroll
  for (int j = 0; j < 8; j++) {
    int i = lane + 32 * j;
    hh[j] = (t[j] - m) * inv * n2g[i] + n2b[i] + xr[i];
    H2[(size_t)row * C + i] = hh[j];
    s += hh[j];
  }
  m = wave_sum32(s) * (1.f / 256.f);
  v = 0.f;
#pragma unroll
  for (int j = 0; j < 8; j++) { float d = hh[j] - m; v += d * d; }
  inv = rsqrtf(wave_sum32(v) * (1.f / 256.f) + LN_EPS);
#pragma unroll
  for (int j = 0; j < 8; j++) {
    int i = lane + 32 * j;
    Z[(size_t)row * C + i] = (bf16)((hh[j] - m) * inv * g1[i] + b1[i]);
  }
}

// BatchNorm stats: one block per channel, mean/var over M rows.
__global__ __launch_bounds__(256) void bn_stats_kernel(
    const float* __restrict__ Y, int M, int O, float* __restrict__ stats)
{
  int ch = blockIdx.x;
  int t = threadIdx.x;
  float s = 0.f, q = 0.f;
  for (int r = t; r < M; r += blockDim.x) {
    float v = Y[(size_t)r * O + ch];
    s += v; q += v * v;
  }
  __shared__ float ss[256], qq[256];
  ss[t] = s; qq[t] = q;
  __syncthreads();
  for (int o = 128; o > 0; o >>= 1) {
    if (t < o) { ss[t] += ss[t + o]; qq[t] += qq[t + o]; }
    __syncthreads();
  }
  if (t == 0) {
    float mm = ss[0] / M;
    stats[ch] = mm;
    stats[O + ch] = qq[0] / M - mm * mm;
  }
}

__global__ __launch_bounds__(256) void bn_gelu_bf16_kernel(
    const float* __restrict__ Y, const float* __restrict__ stats,
    const float* __restrict__ g, const float* __restrict__ be,
    bf16* __restrict__ out, int M, int O)
{
  size_t tid = (size_t)blockIdx.x * blockDim.x + threadIdx.x;
  if (tid >= (size_t)M * O) return;
  int ch = (int)(tid % O);
  float inv = rsqrtf(stats[O + ch] + LN_EPS);
  float y = (Y[tid] - stats[ch]) * inv * g[ch] + be[ch];
  out[tid] = (bf16)geluf(y);
}

__global__ __launch_bounds__(256) void bn_gelu_f32_kernel(
    const float* __restrict__ Y, const float* __restrict__ stats,
    const float* __restrict__ g, const float* __restrict__ be,
    float* __restrict__ out, int M, int O)
{
  size_t tid = (size_t)blockIdx.x * blockDim.x + threadIdx.x;
  if (tid >= (size_t)M * O) return;
  int ch = (int)(tid % O);
  float inv = rsqrtf(stats[O + ch] + LN_EPS);
  float y = (Y[tid] - stats[ch]) * inv * g[ch] + be[ch];
  out[tid] = geluf(y);
}

// lambda = exp(sum lq1*lk1) - exp(sum lq2*lk2) + LAMBDA_INIT
__global__ __launch_bounds__(32) void lam_kernel(
    const float* __restrict__ lq1, const float* __restrict__ lk1,
    const float* __restrict__ lq2, const float* __restrict__ lk2,
    float* __restrict__ lam)
{
  int t = threadIdx.x;
  float a = (t < 16) ? lq1[t] * lk1[t] : 0.f;
  float b = (t < 16) ? lq2[t] * lk2[t] : 0.f;
  a = wave_sum32(a);
  b = wave_sum32(b);
  if (t == 0) *lam = expf(a) - expf(b) + (0.8f - 0.6f * expf(-1.2f));
}

// Differential attention: one WG per (batch b, head h in 0..7), thread = query n.
__global__ __launch_bounds__(128) void attn_kernel(
    const float* __restrict__ Q, const float* __restrict__ Km,
    const float* __restrict__ V, const float* __restrict__ ue,
    const float* __restrict__ lamp, const float* __restrict__ sg,
    const float* __restrict__ sb, bf16* __restrict__ Aout)
{
  int b = blockIdx.x >> 3;
  int h = blockIdx.x & 7;
  int n = threadIdx.x;
  __shared__ float ks0[128][16];
  __shared__ float ks1[128][16];
  __shared__ float vs[128][32];
  size_t rbase = ((size_t)b * 128 + n) * 256;
#pragma unroll
  for (int d = 0; d < 16; d++) {
    ks0[n][d] = Km[rbase + (2 * h) * 16 + d];
    ks1[n][d] = Km[rbase + (2 * h + 1) * 16 + d];
  }
#pragma unroll
  for (int e = 0; e < 32; e++) vs[n][e] = V[rbase + h * 32 + e];
  float q0[16], q1[16];
#pragma unroll
  for (int d = 0; d < 16; d++) {
    q0[d] = Q[rbase + (2 * h) * 16 + d] * 0.25f;      // HD^-0.5 = 1/4
    q1[d] = Q[rbase + (2 * h + 1) * 16 + d] * 0.25f;
  }
  __syncthreads();
  const size_t ub = (size_t)b * 128 * 128 * 16;
  float mx0 = -1e30f, s0 = 0.f, mx1 = -1e30f, s1 = 0.f;
  for (int m = 0; m < 128; m++) {
    float d0 = 0.f, d1 = 0.f;
#pragma unroll
    for (int d = 0; d < 16; d++) { d0 += q0[d] * ks0[m][d]; d1 += q1[d] * ks1[m][d]; }
    float u0 = ue[ub + ((size_t)m * 128 + n) * 16 + 2 * h];
    float u1 = ue[ub + ((size_t)m * 128 + n) * 16 + 2 * h + 1];
    d0 += (u0 != 0.f) ? u0 : -1e30f;
    d1 += (u1 != 0.f) ? u1 : -1e30f;
    float nm0 = fmaxf(mx0, d0); s0 = s0 * expf(mx0 - nm0) + expf(d0 - nm0); mx0 = nm0;
    float nm1 = fmaxf(mx1, d1); s1 = s1 * expf(mx1 - nm1) + expf(d1 - nm1); mx1 = nm1;
  }
  float lam = *lamp;
  float inv0 = (s0 > 0.f) ? 1.f / s0 : 0.f;
  float inv1 = (s1 > 0.f) ? 1.f / s1 : 0.f;
  float acc[32];
#pragma unroll
  for (int e = 0; e < 32; e++) acc[e] = 0.f;
  for (int m = 0; m < 128; m++) {
    float d0 = 0.f, d1 = 0.f;
#pragma unroll
    for (int d = 0; d < 16; d++) { d0 += q0[d] * ks0[m][d]; d1 += q1[d] * ks1[m][d]; }
    float u0 = ue[ub + ((size_t)m * 128 + n) * 16 + 2 * h];
    float u1 = ue[ub + ((size_t)m * 128 + n) * 16 + 2 * h + 1];
    d0 += (u0 != 0.f) ? u0 : -1e30f;
    d1 += (u1 != 0.f) ? u1 : -1e30f;
    float w = expf(d0 - mx0) * inv0 - lam * expf(d1 - mx1) * inv1;
#pragma unroll
    for (int e = 0; e < 32; e++) acc[e] += w * vs[m][e];
  }
  float mean = 0.f;
#pragma unroll
  for (int e = 0; e < 32; e++) mean += acc[e];
  mean *= (1.f / 32.f);
  float var = 0.f;
#pragma unroll
  for (int e = 0; e < 32; e++) { float d = acc[e] - mean; var += d * d; }
  var *= (1.f / 32.f);
  float inv = rsqrtf(var + LN_EPS);
#pragma unroll
  for (int e = 0; e < 32; e++)
    Aout[rbase + h * 32 + e] = (bf16)((acc[e] - mean) * inv * sg[e] + sb[e]);
}

__global__ __launch_bounds__(256) void add2_kernel(
    const float* __restrict__ a, const float* __restrict__ b,
    float* __restrict__ o, int n)
{
  int t = blockIdx.x * blockDim.x + threadIdx.x;
  if (t < n) o[t] = a[t] + b[t];
}

// Mean over N + 2-layer head, one block per batch element.
__global__ __launch_bounds__(256) void head_kernel(
    const float* __restrict__ x, const float* __restrict__ Wh1,
    const float* __restrict__ bh1, const float* __restrict__ Wh2,
    const float* __restrict__ bh2, float* __restrict__ out)
{
  int b = blockIdx.x;
  int t = threadIdx.x;
  __shared__ float xm[256], red[256];
  float s = 0.f;
  for (int n = 0; n < 128; n++) s += x[((size_t)b * 128 + n) * 256 + t];
  xm[t] = s * (1.f / 128.f);
  __syncthreads();
  float d = bh1[t];
  for (int c = 0; c < 256; c++) d += xm[c] * Wh1[(size_t)t * 256 + c];
  red[t] = geluf(d) * Wh2[t];
  __syncthreads();
  for (int o = 128; o > 0; o >>= 1) {
    if (t < o) red[t] += red[t + o];
    __syncthreads();
  }
  if (t == 0) out[b] = red[0] + bh2[0];
}

// ===========================================================================
// Host orchestration
// ===========================================================================
extern "C" void kernel_launch(void* const* d_in, const int* in_sizes, int n_in,
                              void* d_out, int out_size, void* d_ws, size_t ws_size,
                              hipStream_t stream)
{
  (void)in_sizes; (void)n_in; (void)out_size; (void)ws_size;
  auto F = [&](int i) -> const float* { return (const float*)d_in[i]; };

  // deterministic bump allocator into workspace
  char* wsb = (char*)d_ws;
  size_t off = 0;
  auto alloc = [&](size_t bytes) -> void* {
    void* p = wsb + off;
    off = (off + bytes + 255) & ~(size_t)255;
    return p;
  };

  int*   idx   = (int*)  alloc((size_t)4096 * 32 * 4);
  float* xcur  = (float*)alloc((size_t)4096 * 256 * 4);
  float* h2b   = (float*)alloc((size_t)4096 * 256 * 4);
  float* ue    = (float*)alloc((size_t)32 * 128 * 128 * 16 * 4);
  float* stats = (float*)alloc(2 * 64 * 4);
  float* lam   = (float*)alloc(16 * 4);
  float* qb    = (float*)alloc((size_t)4096 * 256 * 4);
  float* kb2   = (float*)alloc((size_t)4096 * 256 * 4);
  float* vb    = (float*)alloc((size_t)4096 * 256 * 4);
  bf16*  abuf  = (bf16*) alloc((size_t)4096 * 256 * 2);
  bf16*  Abuf  = (bf16*) alloc((size_t)33554432 * 2);   // 64 MB
  bf16*  Hbuf  = (bf16*) alloc((size_t)33554432 * 2);   // 64 MB
  float* Ybuf  = (float*)alloc((size_t)33554432 * 4);   // 128 MB

  auto pack = [&](const float* W, int O, int Kw, int Kp) -> bf16* {
    bf16* dst = (bf16*)alloc((size_t)O * Kp * 2);
    int total = O * Kp;
    pack_w_kernel<<<dim3((unsigned)((total + 255) / 256)), dim3(256), 0, stream>>>(
        W, O, Kw, Kp, dst);
    return dst;
  };
  auto gemm = [&](const bf16* A, const bf16* W, const float* bias, float* Y,
                  int M, int K, int O) {
    if (O % 64 == 0) {
      unsigned blocks = (unsigned)((M / 128) * (O / 64));
      gemm_bf16_wmma<4><<<dim3(blocks), dim3(256), 0, stream>>>(A, W, bias, Y, M, K, O);
    } else {
      unsigned blocks = (unsigned)((M / 128) * (O / 16));
      gemm_bf16_wmma<1><<<dim3(blocks), dim3(256), 0, stream>>>(A, W, bias, Y, M, K, O);
    }
  };
  auto lnrows = [&](const float* Y, const float* g, const float* be, bf16* o,
                    int M, int C, int dogelu) {
    ln_act_rows<<<dim3((unsigned)((M + 7) / 8)), dim3(256), 0, stream>>>(
        Y, g, be, o, M, C, dogelu);
  };

  // ---- parameter index map (setup_inputs insertion order) ----
  // 0 particles, 1 interactions,
  // 2..37   edge: (s,l) -> 2+(s*3+l)*4 + {W,b,g,be}
  // 38..53  inter: l -> 38+l*4 + {W,b,g,be}
  // 54..141 blocks: 54+bi*22 + {n1g,n1b,n2g,n2b,Wq,Wk,Wv,Wo,lq1,lk1,lq2,lk2,
  //                             sg,sb,g1,b1,W1,bb1,g2,b2,W2,bb2}
  // 142..145 head: Wh1,bh1,Wh2,bh2
  static const int EDH[3]  = {64, 128, 256};
  static const int KKH[3]  = {16, 32, 32};
  static const int CINH[3] = {11, 64, 128};

  bf16* ewp[3][3];
  for (int s = 0; s < 3; s++)
    for (int l = 0; l < 3; l++) {
      int Kw = (l == 0) ? 2 * CINH[s] : EDH[s];
      int Kp = (Kw + 31) & ~31;
      ewp[s][l] = pack(F(2 + (s * 3 + l) * 4), EDH[s], Kw, Kp);
    }
  static const int IO[4] = {64, 64, 64, 16};
  static const int IK[4] = {4, 64, 64, 64};
  bf16* iwp[4];
  for (int l = 0; l < 4; l++) {
    int Kp = (IK[l] + 31) & ~31;
    iwp[l] = pack(F(38 + l * 4), IO[l], IK[l], Kp);
  }
  bf16 *wq[4], *wk[4], *wv[4], *wo[4], *w1[4], *w2[4];
  for (int b = 0; b < 4; b++) {
    int bb = 54 + b * 22;
    wq[b] = pack(F(bb + 4),  256, 256, 256);
    wk[b] = pack(F(bb + 5),  256, 256, 256);
    wv[b] = pack(F(bb + 6),  256, 256, 256);
    wo[b] = pack(F(bb + 7),  256, 256, 256);
    w1[b] = pack(F(bb + 16), 1024, 256, 256);
    w2[b] = pack(F(bb + 20), 256, 1024, 1024);
  }

  // ---- EdgeConv stages ----
  const float* xin = F(0);
  for (int s = 0; s < 3; s++) {
    int cin = CINH[s], ed = EDH[s], kk = KKH[s];
    knn_kernel<<<dim3(32), dim3(128), 0, stream>>>(xin, cin, kk, idx);
    int Kp0 = (2 * cin + 31) & ~31;
    int Mr = 4096 * kk;
    size_t totalA = (size_t)Mr * Kp0;
    edge_build_kernel<<<dim3((unsigned)((totalA + 255) / 256)), dim3(256), 0, stream>>>(
        xin, idx, cin, kk, Kp0, Abuf);
    int e0 = 2 + (s * 3 + 0) * 4, e1 = 2 + (s * 3 + 1) * 4, e2 = 2 + (s * 3 + 2) * 4;
    gemm(Abuf, ewp[s][0], F(e0 + 1), Ybuf, Mr, Kp0, ed);
    lnrows(Ybuf, F(e0 + 2), F(e0 + 3), Hbuf, Mr, ed, 1);
    gemm(Hbuf, ewp[s][1], F(e1 + 1), Ybuf, Mr, ed, ed);
    lnrows(Ybuf, F(e1 + 2), F(e1 + 3), Abuf, Mr, ed, 1);
    gemm(Abuf, ewp[s][2], F(e2 + 1), Ybuf, Mr, ed, ed);
    lnrows(Ybuf, F(e2 + 2), F(e2 + 3), Hbuf, Mr, ed, 1);
    int totalM = 4096 * ed;
    mean_k_kernel<<<dim3((unsigned)((totalM + 255) / 256)), dim3(256), 0, stream>>>(
        Hbuf, ed, kk, xcur);
    xin = xcur;
  }

  // ---- interaction encoder ----
  {
    int Mr = 32 * 128 * 128;   // 524288 rows
    int totA = Mr * 32;
    pack_w_kernel<<<dim3((unsigned)((totA + 255) / 256)), dim3(256), 0, stream>>>(
        F(1), Mr, 4, 32, Abuf);  // pad 4 -> 32
    bf16* cur = Abuf;
    bf16* nxt = Hbuf;
    for (int l = 0; l < 4; l++) {
      int Kp = (IK[l] + 31) & ~31;
      int O = IO[l];
      gemm(cur, iwp[l], F(38 + l * 4 + 1), Ybuf, Mr, Kp, O);
      bn_stats_kernel<<<dim3(O), dim3(256), 0, stream>>>(Ybuf, Mr, O, stats);
      size_t tot = (size_t)Mr * O;
      unsigned blks = (unsigned)((tot + 255) / 256);
      if (l < 3) {
        bn_gelu_bf16_kernel<<<dim3(blks), dim3(256), 0, stream>>>(
            Ybuf, stats, F(38 + l * 4 + 2), F(38 + l * 4 + 3), nxt, Mr, O);
        bf16* t = cur; cur = nxt; nxt = t;
      } else {
        bn_gelu_f32_kernel<<<dim3(blks), dim3(256), 0, stream>>>(
            Ybuf, stats, F(38 + l * 4 + 2), F(38 + l * 4 + 3), ue, Mr, O);
      }
    }
  }

  // ---- differential attention blocks ----
  for (int b = 0; b < 4; b++) {
    int bb = 54 + b * 22;
    lam_kernel<<<dim3(1), dim3(32), 0, stream>>>(F(bb + 8), F(bb + 9),
                                                 F(bb + 10), F(bb + 11), lam + b);
    lnrows(xcur, F(bb + 0), F(bb + 1), Hbuf, 4096, 256, 0);
    gemm(Hbuf, wq[b], nullptr, qb,  4096, 256, 256);
    gemm(Hbuf, wk[b], nullptr, kb2, 4096, 256, 256);
    gemm(Hbuf, wv[b], nullptr, vb,  4096, 256, 256);
    attn_kernel<<<dim3(256), dim3(128), 0, stream>>>(
        qb, kb2, vb, ue, lam + b, F(bb + 12), F(bb + 13), abuf);
    gemm(abuf, wo[b], nullptr, Ybuf, 4096, 256, 256);
    ln_res_ln_kernel<<<dim3(512), dim3(256), 0, stream>>>(
        Ybuf, xcur, F(bb + 2), F(bb + 3), F(bb + 14), F(bb + 15), h2b, Hbuf, 4096);
    gemm(Hbuf, w1[b], F(bb + 17), Ybuf, 4096, 256, 1024);
    gelu_ln_rows_kernel<<<dim3(512), dim3(256), 0, stream>>>(
        Ybuf, F(bb + 18), F(bb + 19), Hbuf, 4096, 1024);
    gemm(Hbuf, w2[b], F(bb + 21), Ybuf, 4096, 1024, 256);
    add2_kernel<<<dim3(4096), dim3(256), 0, stream>>>(Ybuf, h2b, xcur, 4096 * 256);
  }

  // ---- head ----
  head_kernel<<<dim3(32), dim3(256), 0, stream>>>(
      xcur, F(142), F(143), F(144), F(145), (float*)d_out);
}
